// GraphAttentionLayer_8340826488872
// MI455X (gfx1250) — matrix operand order repaired and checked by reading, not compiled, not run on hardware
//
#include <hip/hip_runtime.h>
#include <hip/hip_bf16.h>
#include <math.h>

#define DIM 96
#define HEADS 4
#define HEAD_DIM 24
#define SLOPE 0.2f
#define LN_EPS 1e-5f

typedef __attribute__((ext_vector_type(16))) _Float16 v16h;
typedef __attribute__((ext_vector_type(8)))  float    v8f;

// ---------------------------------------------------------------------------
// WMMA GEMM 1: h[N,96] = x[N,96] @ W[96,96]
// One wave (32 lanes) per 16x16 output tile; K looped in chunks of 32.
// A fragment (16x32 f16): lane = M(0..15) | half-wave selects K-subblock.
// B fragment (32x16 f16): lane&15 = N, (lane>>4)*16 + idx = K.
// C fragment (16x16 f32): VGPR r -> M = 8*(lane>>4) + r, N = lane&15.
// ---------------------------------------------------------------------------
__global__ void gat_gemm_xW(const float* __restrict__ x,
                            const float* __restrict__ W,
                            float* __restrict__ h) {
  const int lane  = threadIdx.x;     // 0..31, wave32
  const int g     = lane >> 4;       // half-wave group
  const int n15   = lane & 15;
  const int row   = blockIdx.x * 16 + n15;   // M for A loads
  const int col   = blockIdx.y * 16 + n15;   // N for B loads / C store
  const int kbase = g * 8;

  v8f acc = {};
  for (int kc = 0; kc < DIM; kc += 32) {
    v16h a, b;
    const float* xrow = x + (size_t)row * DIM + kc;
#pragma unroll
    for (int v = 0; v < 8; ++v) {
      // contiguous K pair per VGPR: K = 16g + 2v
      const int K0 = 2 * kbase + 2 * v;
      const float2 p = *(const float2*)(xrow + K0);
      a[2 * v]     = (_Float16)p.x;
      a[2 * v + 1] = (_Float16)p.y;
    }
#pragma unroll
    for (int idx = 0; idx < 16; ++idx) {
      const int K = kc + g * 16 + idx;
      b[idx] = (_Float16)W[(size_t)K * DIM + col];
    }
    acc = __builtin_amdgcn_wmma_f32_16x16x32_f16(false, a, false, b,
                                                 (short)0, acc, false, false);
  }
#pragma unroll
  for (int r = 0; r < 8; ++r) {
    const int m = blockIdx.x * 16 + g * 8 + r;
    h[(size_t)m * DIM + col] = acc[r];
  }
}

// ---------------------------------------------------------------------------
// Per-(node,head) attention scores + init of segment max/sum state.
// ---------------------------------------------------------------------------
__global__ void gat_scores(const float* __restrict__ h,
                           const float* __restrict__ att_src,
                           const float* __restrict__ att_dst,
                           float* __restrict__ asrc, float* __restrict__ adst,
                           float* __restrict__ segmax, float* __restrict__ segsum,
                           int n_nodes) {
  const int t = blockIdx.x * blockDim.x + threadIdx.x;
  if (t >= n_nodes * HEADS) return;
  const int n  = t >> 2;
  const int hd = t & 3;
  const float* hp = h + (size_t)n * DIM + hd * HEAD_DIM;
  const float* as = att_src + hd * HEAD_DIM;
  const float* ad = att_dst + hd * HEAD_DIM;
  float ss = 0.f, sd = 0.f;
#pragma unroll
  for (int k = 0; k < HEAD_DIM; ++k) {
    const float v = hp[k];
    ss += v * as[k];
    sd += v * ad[k];
  }
  asrc[t]   = ss;
  adst[t]   = sd;
  segmax[t] = -1e30f;
  segsum[t] = 0.f;
}

__global__ void gat_zero(float* __restrict__ p, int n) {
  const int t = blockIdx.x * blockDim.x + threadIdx.x;
  if (t < n) p[t] = 0.f;
}

// sign-split monotonic float atomic max (init must be negative)
__device__ inline void atomicMaxF32(float* addr, float val) {
  if (val >= 0.f) atomicMax((int*)addr, __float_as_int(val));
  else            atomicMin((unsigned int*)addr, __float_as_uint(val));
}

__device__ inline int edge_src(const long long* ei, int e, int nE) {
  return (e < nE) ? (int)ei[e] : (e - nE);
}
__device__ inline int edge_dst(const long long* ei, int e, int nE) {
  return (e < nE) ? (int)ei[(size_t)nE + e] : (e - nE);
}

// ---------------------------------------------------------------------------
// Pass 1: segment max of leaky-relu logits over incoming edges of each dst.
// ---------------------------------------------------------------------------
__global__ void gat_edge_max(const long long* __restrict__ ei,
                             const float* __restrict__ asrc,
                             const float* __restrict__ adst,
                             float* __restrict__ segmax,
                             int nE, int n_nodes) {
  const int t = blockIdx.x * blockDim.x + threadIdx.x;
  const int total = (nE + n_nodes) * HEADS;
  if (t >= total) return;
  const int e  = t >> 2;
  const int hd = t & 3;
  const int s = edge_src(ei, e, nE);
  const int d = edge_dst(ei, e, nE);
  float v = asrc[s * HEADS + hd] + adst[d * HEADS + hd];
  v = (v > 0.f) ? v : SLOPE * v;
  atomicMaxF32(&segmax[d * HEADS + hd], v);
}

// ---------------------------------------------------------------------------
// Pass 2: w = exp(e - max), store per-edge weight, segment-sum denominator.
// ---------------------------------------------------------------------------
__global__ void gat_edge_exp(const long long* __restrict__ ei,
                             const float* __restrict__ asrc,
                             const float* __restrict__ adst,
                             const float* __restrict__ segmax,
                             float* __restrict__ segsum,
                             float* __restrict__ ew,
                             int nE, int n_nodes) {
  const int t = blockIdx.x * blockDim.x + threadIdx.x;
  const int total = (nE + n_nodes) * HEADS;
  if (t >= total) return;
  const int e  = t >> 2;
  const int hd = t & 3;
  const int s = edge_src(ei, e, nE);
  const int d = edge_dst(ei, e, nE);
  float v = asrc[s * HEADS + hd] + adst[d * HEADS + hd];
  v = (v > 0.f) ? v : SLOPE * v;
  const float w = __expf(v - segmax[d * HEADS + hd]);
  ew[t] = w;
  atomicAdd(&segsum[d * HEADS + hd], w);
}

// ---------------------------------------------------------------------------
// Pass 3: weighted scatter-aggregate. One wave per edge; lane covers 3 feats.
// h + accum + softmax state are L2-resident (<40 MB of 192 MB L2), so the
// random gather + float atomics run at L2 bandwidth, not HBM.
// ---------------------------------------------------------------------------
__global__ void gat_edge_aggregate(const long long* __restrict__ ei,
                                   const float* __restrict__ h,
                                   const float* __restrict__ ew,
                                   const float* __restrict__ segsum,
                                   float* __restrict__ accum,
                                   int nE, int n_nodes) {
  const int warp = (blockIdx.x * blockDim.x + threadIdx.x) >> 5;
  const int lane = threadIdx.x & 31;
  const int total = nE + n_nodes;
  if (warp >= total) return;
  const int e = warp;
  const int s = edge_src(ei, e, nE);
  const int d = edge_dst(ei, e, nE);
  const float* hs = h + (size_t)s * DIM;
  __builtin_prefetch(hs + lane, 0, 0);   // global_prefetch_b8
#pragma unroll
  for (int i = 0; i < 3; ++i) {
    const int f  = lane + 32 * i;
    const int hd = f / HEAD_DIM;
    const float alpha = ew[e * HEADS + hd] / segsum[d * HEADS + hd];
    atomicAdd(&accum[(size_t)d * DIM + f], alpha * hs[f]);
  }
}

// ---------------------------------------------------------------------------
// WMMA GEMM 2: proj[N,96] = (accum + gat_bias) @ proj_W^T + proj_b
// B[k][j] = proj_W[j][k]  (row of proj_W is contiguous in k).
// ---------------------------------------------------------------------------
__global__ void gat_gemm_proj(const float* __restrict__ accum,
                              const float* __restrict__ gat_bias,
                              const float* __restrict__ projW,
                              const float* __restrict__ projb,
                              float* __restrict__ proj) {
  const int lane  = threadIdx.x;
  const int g     = lane >> 4;
  const int n15   = lane & 15;
  const int row   = blockIdx.x * 16 + n15;
  const int col   = blockIdx.y * 16 + n15;
  const int kbase = g * 8;

  v8f acc = {};
  for (int kc = 0; kc < DIM; kc += 32) {
    v16h a, b;
    const float* arow = accum + (size_t)row * DIM + kc;
    const float* brow = gat_bias + kc;
#pragma unroll
    for (int v = 0; v < 8; ++v) {
      const int K0 = 2 * kbase + 2 * v;
      const float2 p = *(const float2*)(arow + K0);
      a[2 * v]     = (_Float16)(p.x + brow[K0]);
      a[2 * v + 1] = (_Float16)(p.y + brow[K0 + 1]);
    }
    const float* wrow = projW + (size_t)col * DIM + kc + g * 16;
#pragma unroll
    for (int idx = 0; idx < 16; ++idx) {
      b[idx] = (_Float16)wrow[idx];
    }
    acc = __builtin_amdgcn_wmma_f32_16x16x32_f16(false, a, false, b,
                                                 (short)0, acc, false, false);
  }
  const float bias = projb[col];
#pragma unroll
  for (int r = 0; r < 8; ++r) {
    const int m = blockIdx.x * 16 + g * 8 + r;
    proj[(size_t)m * DIM + col] = acc[r] + bias;
  }
}

// ---------------------------------------------------------------------------
// Residual + LayerNorm: one wave32 per node, shuffle reduction (width 32).
// ---------------------------------------------------------------------------
__global__ void gat_residual_ln(const float* __restrict__ x,
                                const float* __restrict__ proj,
                                const float* __restrict__ lng,
                                const float* __restrict__ lnb,
                                float* __restrict__ out, int n_nodes) {
  const int warp = (blockIdx.x * blockDim.x + threadIdx.x) >> 5;
  const int lane = threadIdx.x & 31;
  if (warp >= n_nodes) return;
  const size_t base = (size_t)warp * DIM;
  float z[3];
  float s = 0.f, s2 = 0.f;
#pragma unroll
  for (int i = 0; i < 3; ++i) {
    const int f = lane + 32 * i;
    z[i] = x[base + f] + proj[base + f];
    s  += z[i];
    s2 += z[i] * z[i];
  }
#pragma unroll
  for (int off = 16; off > 0; off >>= 1) {
    s  += __shfl_xor(s,  off, 32);
    s2 += __shfl_xor(s2, off, 32);
  }
  const float mu  = s / (float)DIM;
  const float var = s2 / (float)DIM - mu * mu;
  const float inv = rsqrtf(var + LN_EPS);
#pragma unroll
  for (int i = 0; i < 3; ++i) {
    const int f = lane + 32 * i;
    out[base + f] = lng[f] * (z[i] - mu) * inv + lnb[f];
  }
}

// ---------------------------------------------------------------------------
extern "C" void kernel_launch(void* const* d_in, const int* in_sizes, int n_in,
                              void* d_out, int out_size, void* d_ws, size_t ws_size,
                              hipStream_t stream) {
  const float*     x        = (const float*)d_in[0];
  const long long* ei       = (const long long*)d_in[1];  // jnp.int64
  const float*     W        = (const float*)d_in[2];
  const float*     att_src  = (const float*)d_in[3];
  const float*     att_dst  = (const float*)d_in[4];
  const float*     gat_bias = (const float*)d_in[5];
  const float*     projW    = (const float*)d_in[6];
  const float*     projb    = (const float*)d_in[7];
  const float*     lng      = (const float*)d_in[8];
  const float*     lnb      = (const float*)d_in[9];
  float*           out      = (float*)d_out;

  const int n_nodes = in_sizes[0] / DIM;   // 50000 (divisible by 16)
  const int nE      = in_sizes[1] / 2;     // 800000
  const int EN      = nE + n_nodes;

  // workspace layout (floats)
  float* ws     = (float*)d_ws;
  float* h      = ws;                                      // N*96
  float* asrc   = h      + (size_t)n_nodes * DIM;          // N*4
  float* adst   = asrc   + (size_t)n_nodes * HEADS;        // N*4
  float* segmax = adst   + (size_t)n_nodes * HEADS;        // N*4
  float* segsum = segmax + (size_t)n_nodes * HEADS;        // N*4
  float* ew     = segsum + (size_t)n_nodes * HEADS;        // (E+N)*4
  float* accum  = ew     + (size_t)EN * HEADS;             // N*96
  float* proj   = h;  // h is dead after aggregation; reuse for proj output

  const dim3 gemm_grid(n_nodes / 16, DIM / 16);

  gat_gemm_xW<<<gemm_grid, 32, 0, stream>>>(x, W, h);

  gat_scores<<<(n_nodes * HEADS + 255) / 256, 256, 0, stream>>>(
      h, att_src, att_dst, asrc, adst, segmax, segsum, n_nodes);

  gat_zero<<<(n_nodes * DIM + 255) / 256, 256, 0, stream>>>(accum, n_nodes * DIM);

  gat_edge_max<<<(EN * HEADS + 255) / 256, 256, 0, stream>>>(
      ei, asrc, adst, segmax, nE, n_nodes);

  gat_edge_exp<<<(EN * HEADS + 255) / 256, 256, 0, stream>>>(
      ei, asrc, adst, segmax, segsum, ew, nE, n_nodes);

  gat_edge_aggregate<<<(EN * 32 + 255) / 256, 256, 0, stream>>>(
      ei, h, ew, segsum, accum, nE, n_nodes);

  gat_gemm_proj<<<gemm_grid, 32, 0, stream>>>(accum, gat_bias, projW, projb, proj);

  gat_residual_ln<<<(n_nodes * 32 + 255) / 256, 256, 0, stream>>>(
      x, proj, lng, lnb, out, n_nodes);
}
